// TokenSampleRelation_48524540510926
// MI455X (gfx1250) — matrix operand order repaired and checked by reading, not compiled, not run on hardware
//
#include <hip/hip_runtime.h>
#include <hip/hip_bf16.h>
#include <math.h>

// ---------------------------------------------------------------------------
// TokenSampleRelation forward for gfx1250 (MI455X).
// All GEMMs / convs run through v_wmma_f32_16x16x32_bf16 (f32 accum).
// Wave tile = 16x64 (4 WMMAs per K-chunk, A-fragment reused 4x).
// Block = 128 threads = 4 waves stacked in M; B tile (32x64) shared in LDS.
// LDS holds packed bf16 pairs (cvt_pk at fill, ds_load_b32 per fragment pair).
// ---------------------------------------------------------------------------

#define DEV __device__ __forceinline__

typedef __bf16 bf16_t;
typedef bf16_t v16bf  __attribute__((ext_vector_type(16)));
typedef bf16_t bf16x2 __attribute__((ext_vector_type(2)));
typedef float  v8f    __attribute__((ext_vector_type(8)));
static_assert(sizeof(v16bf) == 32, "bf16x16");
static_assert(sizeof(v8f) == 32, "f32x8");

union FragBF { v16bf v; bf16x2 p[8]; };

DEV bf16x2 pack2(float x, float y) {
  bf16x2 r;
  r.x = (bf16_t)x;
  r.y = (bf16_t)y;
  return r;
}

// Problem constants
constexpr int Bq  = 2;
constexpr int Cq  = 256;
constexpr int Hq  = 120;
constexpr int Wq  = 160;
constexpr int Lq  = 50;
constexpr int HWq = Hq * Wq;      // 19200
constexpr int SPq = 150 * 100;    // 15000 spatial of samp_rel

DEV float geluf(float x) { return 0.5f * x * (1.0f + erff(x * 0.70710678118654752f)); }

// ---------------------------------------------------------------------------
// WMMA fragment layouts (CDNA5 ISA §7.12.2), bf16 pairs:
//   A (16x32, MxK): row = lane&15 (both halves); VGPR v holds K pair
//                   pairIdx = (v&3) + 8*(v>>2) + 4*(lane>>4)
//   B (32x16, KxN): col = lane&15; VGPR v holds K pair pairIdx = (v&3) + 8*(v>>2) + 4*(lane>>4)
//   D (16x16):      VGPR r: M = r + 8*(lane>>4), N = lane&15
// ---------------------------------------------------------------------------

// ---------------------------------------------------------------------------
// Generic GEMM  D[M,N] = act(A[M,K] @ B[K,N] + bias)
// AMODE: 0 row-major A[m*K+k]
//        1 "planes"  A[((m/P)*K + k)*P + m%P]   (+ optional fused BN+ReLU)
//        2 stride-6 im2col for g_w2 conv (reads g1 planes (B,256,300,50))
// BMODE: 0 row-major B[k*N+n], 1 transposed B[n*K+k]  (Cout-major weights)
// DMODE: 0 row-major, 1 planes D[((m/DP)*N+n)*DP + m%DP],
//        2 final transpose(0,3,1,2) pack G2[(b*50+l2)*12800 + n*50 + l]
// ACT:   0 none, 1 gelu, 2 sigmoid
// ---------------------------------------------------------------------------
struct GemmArgs {
  const float* A;
  const float* Bw;
  const float* bias;   // may be null
  const float* bnS;    // may be null (AMODE 1 only): relu(v*bnS[k]*bnScale+bnB[k])
  const float* bnB;
  float bnScale;
  float* D;
  int M, N, K, P, DP;
};

template <int DM>
DEV void storeD(const GemmArgs& a, int m, int n, float v) {
  if (DM == 0) {
    a.D[(size_t)m * a.N + n] = v;
  } else if (DM == 1) {
    a.D[((size_t)(m / a.DP) * a.N + n) * a.DP + (m % a.DP)] = v;
  } else {  // DM == 2: g (B,Cout,l,l2) -> G2[b, l2, n*50 + l]
    const int b = m / 2500, pos = m % 2500;
    const int l = pos / 50, l2 = pos % 50;
    a.D[(size_t)(b * 50 + l2) * 12800 + (size_t)n * 50 + l] = v;
  }
}

template <int AM, int BM, int DM, int ACT>
__global__ __launch_bounds__(128) void k_gemm(GemmArgs a) {
  __shared__ bf16x2 lA[4][16][17];   // per-wave A tile  (16 m x 16 K-pairs)
  __shared__ bf16x2 lB[16][72];      // shared  B tile  (16 K-pairs x 64 n)
  const int tid  = threadIdx.x;
  const int lane = tid & 31;
  const int wv   = tid >> 5;
  const int hf   = lane >> 4;
  const int nl   = lane & 15;
  const int m0   = ((int)blockIdx.y << 6) + (wv << 4);
  const int n0   = (int)blockIdx.x << 6;

  // ---- per-lane A row precompute (hoisted out of the K loop) ----
  const int  rA   = nl;                   // row within wave tile
  const int  pbA  = hf << 3;              // this lane fills K-pairs pbA..pbA+7
  const int  mA   = m0 + rA;
  const bool mOk  = (mA < a.M);
  const int  mAc  = mOk ? mA : 0;
  const float* aRow = nullptr;
  int gb = 0, gl = 0, gl2 = 0;
  if (AM == 0) {
    aRow = a.A + (size_t)mAc * a.K;                       // element k: aRow[k]
  } else if (AM == 1) {
    const int q = mAc / a.P, rm = mAc % a.P;
    aRow = a.A + (size_t)q * a.K * a.P + rm;              // element k: aRow[k*P]
  } else {
    const int pos = mAc % 2500;
    gb = mAc / 2500; gl = pos / 50; gl2 = pos % 50;
  }

  // ---- per-thread B column precompute ----
  const int  nB  = tid & 63;              // column within block tile
  const int  kpB = (tid >> 6) << 3;       // this thread fills K-pairs kpB..kpB+7
  const int  nG  = n0 + nB;
  const bool nOk = (nG < a.N);
  const float* bCol = (BM == 0) ? (a.Bw + (nOk ? nG : 0))
                                : (a.Bw + (size_t)(nOk ? nG : 0) * a.K);

  v8f acc0 = {}, acc1 = {}, acc2 = {}, acc3 = {};

  for (int k0 = 0; k0 < a.K; k0 += 32) {
    // ---- fill A tile (this wave's 16x32 slice) ----
#pragma unroll
    for (int i = 0; i < 8; ++i) {
      const int kp = pbA + i;
      const int k  = k0 + (kp << 1);
      float va = 0.0f, vb = 0.0f;
      if (mOk) {
        if (AM == 0) {
          if (k < a.K)     va = aRow[k];
          if (k + 1 < a.K) vb = aRow[k + 1];
        } else if (AM == 1) {
          if (k < a.K)     va = aRow[(size_t)k * a.P];
          if (k + 1 < a.K) vb = aRow[(size_t)(k + 1) * a.P];
          if (a.bnS) {
            if (k < a.K)     va = fmaxf(va * (a.bnS[k] * a.bnScale) + a.bnB[k], 0.0f);
            if (k + 1 < a.K) vb = fmaxf(vb * (a.bnS[k + 1] * a.bnScale) + a.bnB[k + 1], 0.0f);
          }
        } else {  // AM == 2: k = ci*6 + r -> g1[b, ci, gl*6+r, gl2]
          if (k < a.K) {
            const int ci = k / 6, r6 = k - 6 * ci;
            va = a.A[(size_t)(gb * 256 + ci) * 15000 + (size_t)(gl * 6 + r6) * 50 + gl2];
          }
          if (k + 1 < a.K) {
            const int ci = (k + 1) / 6, r6 = (k + 1) - 6 * ci;
            vb = a.A[(size_t)(gb * 256 + ci) * 15000 + (size_t)(gl * 6 + r6) * 50 + gl2];
          }
        }
      }
      lA[wv][rA][kp] = pack2(va, vb);
    }
    // ---- fill shared B tile (32x64) ----
#pragma unroll
    for (int i = 0; i < 8; ++i) {
      const int kp = kpB + i;
      const int k  = k0 + (kp << 1);
      float va = 0.0f, vb = 0.0f;
      if (nOk) {
        if (BM == 0) {
          if (k < a.K)     va = bCol[(size_t)k * a.N];
          if (k + 1 < a.K) vb = bCol[(size_t)(k + 1) * a.N];
        } else {
          if (k < a.K)     va = bCol[k];
          if (k + 1 < a.K) vb = bCol[k + 1];
        }
      }
      lB[kp][nB] = pack2(va, vb);
    }
    if (BM == 0 && k0 + 32 < a.K)   // global_prefetch_b8 next weight chunk
      __builtin_prefetch(bCol + (size_t)(k0 + 32) * a.N, 0, 1);
    __syncthreads();

    // ---- fragments + 4 WMMAs (A fragment reused across the 4 N sub-tiles) --
    FragBF af;
#pragma unroll
    for (int v = 0; v < 8; ++v)
      af.p[v] = lA[wv][nl][(v & 3) + ((v >> 2) << 3) + (hf << 2)];
    FragBF b0, b1, b2, b3;
#pragma unroll
    for (int v = 0; v < 8; ++v) {
      const int kp = (v & 3) + ((v >> 2) << 3) + (hf << 2);
      b0.p[v] = lB[kp][nl];
      b1.p[v] = lB[kp][nl + 16];
      b2.p[v] = lB[kp][nl + 32];
      b3.p[v] = lB[kp][nl + 48];
    }
    acc0 = __builtin_amdgcn_wmma_f32_16x16x32_bf16(false, af.v, false, b0.v, (short)0, acc0, false, false);
    acc1 = __builtin_amdgcn_wmma_f32_16x16x32_bf16(false, af.v, false, b1.v, (short)0, acc1, false, false);
    acc2 = __builtin_amdgcn_wmma_f32_16x16x32_bf16(false, af.v, false, b2.v, (short)0, acc2, false, false);
    acc3 = __builtin_amdgcn_wmma_f32_16x16x32_bf16(false, af.v, false, b3.v, (short)0, acc3, false, false);
    __syncthreads();
  }

  // ---- epilogue: D VGPR r -> M = r + 8*hf, N = n0 + 16*t + nl ----
#pragma unroll
  for (int r = 0; r < 8; ++r) {
    const int m = m0 + (hf << 3) + r;
    if (m >= a.M) continue;
#pragma unroll
    for (int t = 0; t < 4; ++t) {
      const int n = n0 + (t << 4) + nl;
      if (n >= a.N) continue;
      float v = (t == 0 ? acc0[r] : t == 1 ? acc1[r] : t == 2 ? acc2[r] : acc3[r]);
      v += (a.bias ? a.bias[n] : 0.0f);
      if (ACT == 1) v = geluf(v);
      else if (ACT == 2) v = 1.0f / (1.0f + expf(-v));
      storeD<DM>(a, m, n, v);
    }
  }
}

// ---------------------------------------------------------------------------
// Implicit-GEMM 3x3 (dilated) conv, zero padding = dil.
//   Dst[b,co,p] (+)= act( sum_{ci,tap} tr(X[b,ci,y+dy,x+dx]) * Wt[co,ci,ty,tx] + bias )
//   tr = optional relu(bn).  K = tap*Cin + ci; Cin%32==0 so tap is chunk-uniform.
// ---------------------------------------------------------------------------
struct ConvArgs {
  const float* X;      // (Bb, Cin, Hh, Ww)
  const float* Wt;     // (Cout, Cin, 3, 3)
  const float* bnS;    // optional
  const float* bnB;
  float bnScale;
  const float* bias;   // optional
  float* Dst;          // (Bb, Cout, Hh*Ww)
  int Cin, Cout, Hh, Ww, dil, Bb, acc, act;
};

__global__ __launch_bounds__(128) void k_conv3x3(ConvArgs c) {
  __shared__ bf16x2 lA[4][16][17];
  __shared__ bf16x2 lB[16][72];
  const int tid  = threadIdx.x;
  const int lane = tid & 31;
  const int wv   = tid >> 5;
  const int hf   = lane >> 4;
  const int nl   = lane & 15;
  const int HW   = c.Hh * c.Ww;
  const int M    = c.Bb * HW;
  const int K    = 9 * c.Cin;
  const int m0   = ((int)blockIdx.y << 6) + (wv << 4);
  const int n0   = (int)blockIdx.x << 6;

  // per-lane output-position precompute (hoisted)
  const int  rA  = nl;
  const int  pbA = hf << 3;
  const int  mA  = m0 + rA;
  const bool mOk = (mA < M);
  const int  mAc = mOk ? mA : 0;
  const int  pb_ = mAc / HW;            // batch
  const int  pp  = mAc % HW;
  const int  y0  = pp / c.Ww, x0 = pp % c.Ww;

  // per-thread weight-column precompute
  const int  nB  = tid & 63;
  const int  kpB = (tid >> 6) << 3;
  const int  nG  = n0 + nB;
  const bool nOk = (nG < c.Cout);
  const float* wBase0 = c.Wt + (size_t)(nOk ? nG : 0) * c.Cin * 9;

  v8f acc0 = {}, acc1 = {}, acc2 = {}, acc3 = {};

  for (int k0 = 0; k0 < K; k0 += 32) {
    // chunk-uniform tap decomposition (Cin % 32 == 0)
    const int tap = k0 / c.Cin;
    const int ci0 = k0 - tap * c.Cin;
    const int ty  = tap / 3, tx = tap - 3 * ty;
    const int y   = y0 + (ty - 1) * c.dil;
    const int x   = x0 + (tx - 1) * c.dil;
    const bool inb = mOk && (y >= 0) && (y < c.Hh) && (x >= 0) && (x < c.Ww);
    const float* src = c.X + ((size_t)(pb_ * c.Cin + ci0) * c.Hh + y) * c.Ww + x;

    // ---- fill A (im2col) ----
#pragma unroll
    for (int i = 0; i < 8; ++i) {
      const int kp  = pbA + i;
      const int cia = ci0 + (kp << 1);
      float va = 0.0f, vb = 0.0f;
      if (inb) {
        va = src[(size_t)(kp << 1) * HW];
        vb = src[(size_t)((kp << 1) + 1) * HW];
        if (c.bnS) {
          va = fmaxf(va * (c.bnS[cia] * c.bnScale) + c.bnB[cia], 0.0f);
          vb = fmaxf(vb * (c.bnS[cia + 1] * c.bnScale) + c.bnB[cia + 1], 0.0f);
        }
      }
      lA[wv][rA][kp] = pack2(va, vb);
    }
    // ---- fill B (weights, Cout-major): Wt[(n*Cin+ci)*9 + tap] ----
#pragma unroll
    for (int i = 0; i < 8; ++i) {
      const int kp  = kpB + i;
      const int cia = ci0 + (kp << 1);
      float va = 0.0f, vb = 0.0f;
      if (nOk) {
        va = wBase0[(size_t)cia * 9 + tap];
        vb = wBase0[(size_t)(cia + 1) * 9 + tap];
      }
      lB[kp][nB] = pack2(va, vb);
    }
    __syncthreads();

    FragBF af;
#pragma unroll
    for (int v = 0; v < 8; ++v)
      af.p[v] = lA[wv][nl][(v & 3) + ((v >> 2) << 3) + (hf << 2)];
    FragBF b0, b1, b2, b3;
#pragma unroll
    for (int v = 0; v < 8; ++v) {
      const int kp = (v & 3) + ((v >> 2) << 3) + (hf << 2);
      b0.p[v] = lB[kp][nl];
      b1.p[v] = lB[kp][nl + 16];
      b2.p[v] = lB[kp][nl + 32];
      b3.p[v] = lB[kp][nl + 48];
    }
    acc0 = __builtin_amdgcn_wmma_f32_16x16x32_bf16(false, af.v, false, b0.v, (short)0, acc0, false, false);
    acc1 = __builtin_amdgcn_wmma_f32_16x16x32_bf16(false, af.v, false, b1.v, (short)0, acc1, false, false);
    acc2 = __builtin_amdgcn_wmma_f32_16x16x32_bf16(false, af.v, false, b2.v, (short)0, acc2, false, false);
    acc3 = __builtin_amdgcn_wmma_f32_16x16x32_bf16(false, af.v, false, b3.v, (short)0, acc3, false, false);
    __syncthreads();
  }

#pragma unroll
  for (int r = 0; r < 8; ++r) {
    const int m = m0 + (hf << 3) + r;
    if (m >= M) continue;
    const int mq = m / HW, mr = m - mq * HW;   // hoisted per row
#pragma unroll
    for (int t = 0; t < 4; ++t) {
      const int n = n0 + (t << 4) + nl;
      if (n >= c.Cout) continue;
      float v = (t == 0 ? acc0[r] : t == 1 ? acc1[r] : t == 2 ? acc2[r] : acc3[r]);
      v += (c.bias ? c.bias[n] : 0.0f);
      if (c.act == 1) v = geluf(v);
      const size_t o = ((size_t)mq * c.Cout + n) * HW + mr;
      if (c.acc) c.Dst[o] += v; else c.Dst[o] = v;
    }
  }
}

// ---------------------------------------------------------------------------
// Elementwise / sampling kernels
// ---------------------------------------------------------------------------

__global__ void k_sample_nearest(const float* depth, const float* gpos,
                                 const float* refer, const float* coarse,
                                 const float* oldd, float* sampD, float* coarseD) {
  const int idx = blockIdx.x * blockDim.x + threadIdx.x;
  if (idx >= Bq * Cq * 150) return;
  const int j = idx % 150;
  const int c = (idx / 150) % Cq;
  const int b = idx / (150 * Cq);
  float gx, gy;
  int lcoarse = 0;
  if (j < 100) {
    const int l = j >> 1, t = j & 1;
    const float* g = refer + ((size_t)(b * Lq + l) * 2 + t) * 2;
    gx = g[0]; gy = g[1];
  } else {
    lcoarse = j - 100;
    const float* g = coarse + (size_t)(b * Lq + lcoarse) * 2;
    gx = g[0]; gy = g[1];
  }
  const float fx = ((gx + 1.0f) * (float)Wq - 1.0f) * 0.5f;
  const float fy = ((gy + 1.0f) * (float)Hq - 1.0f) * 0.5f;
  const float rx = rintf(fx), ry = rintf(fy);   // jnp.round == RNE
  const bool valid = (rx >= 0.0f) && (rx < (float)Wq) && (ry >= 0.0f) && (ry < (float)Hq);
  const int ix = (int)fminf(fmaxf(rx, 0.0f), (float)(Wq - 1));
  const int iy = (int)fminf(fmaxf(ry, 0.0f), (float)(Hq - 1));
  const size_t off = ((size_t)(b * Cq + c) * Hq + iy) * Wq + ix;
  const float v = valid ? (depth[off] + gpos[off]) : 0.0f;
  if (j < 100) sampD[(size_t)(b * Cq + c) * 100 + j] = v;
  else         coarseD[(size_t)(b * Cq + c) * 50 + lcoarse] = v + oldd[b * Lq + lcoarse];
}

__global__ void k_sample_bilinear(const float* segmap, const float* refer,
                                  const float* coarse, float* sampSeg) {
  const int idx = blockIdx.x * blockDim.x + threadIdx.x;
  if (idx >= Bq * Cq * 150) return;
  const int i = idx % 150;
  const int c = (idx / 150) % Cq;
  const int b = idx / (150 * Cq);
  const int l = i / 3, t = i % 3;
  float gx, gy;
  if (t < 2) {
    const float* g = refer + ((size_t)(b * Lq + l) * 2 + t) * 2;
    gx = g[0]; gy = g[1];
  } else {
    const float* g = coarse + (size_t)(b * Lq + l) * 2;
    gx = g[0]; gy = g[1];
  }
  const float fx = ((gx + 1.0f) * (float)Wq - 1.0f) * 0.5f;
  const float fy = ((gy + 1.0f) * (float)Hq - 1.0f) * 0.5f;
  const float x0 = floorf(fx), y0 = floorf(fy);
  const float wx = fx - x0, wy = fy - y0;
  const float* img = segmap + (size_t)(b * Cq + c) * HWq;
  auto tap = [&](float yy, float xx, float wgt) -> float {
    const bool valid = (xx >= 0.0f) && (xx < (float)Wq) && (yy >= 0.0f) && (yy < (float)Hq);
    if (!valid) return 0.0f;
    const int xi = (int)fminf(fmaxf(xx, 0.0f), (float)(Wq - 1));
    const int yi = (int)fminf(fmaxf(yy, 0.0f), (float)(Hq - 1));
    return img[(size_t)yi * Wq + xi] * wgt;
  };
  const float v = tap(y0, x0, (1.0f - wy) * (1.0f - wx)) +
                  tap(y0, x0 + 1.0f, (1.0f - wy) * wx) +
                  tap(y0 + 1.0f, x0, wy * (1.0f - wx)) +
                  tap(y0 + 1.0f, x0 + 1.0f, wy * wx);
  sampSeg[(size_t)(b * Cq + c) * 150 + i] = v;
}

__global__ void k_samp_rel(const float* sseg, const float* sdep, float* out) {
  const int idx = blockIdx.x * blockDim.x + threadIdx.x;
  if (idx >= Bq * Cq * SPq) return;
  const int j  = idx % 100;
  const int i  = (idx / 100) % 150;
  const int bc = idx / SPq;
  out[idx] = sseg[(size_t)bc * 150 + i] * sdep[(size_t)bc * 100 + j];
}

__global__ void k_geometry(const float* refer, const float* coarse, float* geo) {
  const int idx = blockIdx.x * blockDim.x + threadIdx.x;
  if (idx >= Bq * Lq * 3 * Lq) return;
  const int l2 = idx % 50;
  const int s  = (idx / 50) % 3;
  const int l  = (idx / 150) % 50;
  const int b  = idx / 7500;
  float v;
  if (s < 2) {
    const float* r = refer + ((size_t)(b * Lq + l) * 2 + s) * 2;
    const float* cp = coarse + (size_t)(b * Lq + l2) * 2;
    const float dx = r[0] - cp[0], dy = r[1] - cp[1];
    v = sqrtf(dx * dx + dy * dy);
  } else {
    const float* cp = coarse + (size_t)(b * Lq + l) * 2;
    const float* p1 = refer + ((size_t)(b * Lq + l2) * 2 + 0) * 2;
    const float* p2 = refer + ((size_t)(b * Lq + l2) * 2 + 1) * 2;
    v = (cp[0] - p1[0]) * (p2[1] - p1[1]) - (cp[1] - p1[1]) * (p2[0] - p1[0]);
  }
  geo[idx] = v;
}

__global__ void k_rela_geom(const float* rela, const float* geo,
                            const float* cdp, float* rg) {
  const int idx = blockIdx.x * blockDim.x + threadIdx.x;
  if (idx >= Bq * Cq * 300 * 50) return;
  const int l2 = idx % 50;
  const int q  = (idx / 50) % 300;
  const int bc = idx / 15000;
  const int b  = bc >> 8;
  const int l = q / 6, s = q % 6;
  float v;
  if (s < 2)      v = rela[(size_t)bc * 5000 + (size_t)l2 * 100 + 2 * l + s];
  else if (s < 5) v = geo[((size_t)(b * Lq + l) * 3 + (s - 2)) * 50 + l2];
  else            v = cdp[(size_t)bc * 50 + l2];
  rg[idx] = v;
}

// ---------------------------------------------------------------------------
// Host-side launch helpers
// ---------------------------------------------------------------------------
template <int AM, int BM, int DM, int ACT>
static void gemm(hipStream_t s, const float* A, const float* Bw, const float* bias,
                 const float* bnS, const float* bnB, float bnScale, float* D,
                 int M, int N, int K, int P, int DP) {
  GemmArgs a{A, Bw, bias, bnS, bnB, bnScale, D, M, N, K, P, DP};
  dim3 grd((unsigned)((N + 63) / 64), (unsigned)((M + 63) / 64));
  k_gemm<AM, BM, DM, ACT><<<grd, dim3(128), 0, s>>>(a);
}

static void conv3x3(hipStream_t s, const float* X, const float* Wt,
                    const float* bnS, const float* bnB, float bnScale,
                    const float* bias, float* Dst,
                    int Cin, int Cout, int Hh, int Ww, int dil,
                    int Bb, int acc, int act) {
  ConvArgs c{X, Wt, bnS, bnB, bnScale, bias, Dst, Cin, Cout, Hh, Ww, dil, Bb, acc, act};
  const int M = Bb * Hh * Ww;
  dim3 grd((unsigned)((Cout + 63) / 64), (unsigned)((M + 63) / 64));
  k_conv3x3<<<grd, dim3(128), 0, s>>>(c);
}

// Input indices: setup_inputs() dict order, params flattened depth-first.
enum {
  IN_SEG = 0, IN_DEPTH, IN_REFER, IN_COARSE, IN_GPOS, IN_OLDD,
  P_SEG_W1, P_SEG_B1, P_SEG_W2, P_SEG_B2,
  P_CD_W1, P_CD_B1, P_CD_W2, P_CD_B2,
  P_A1_BN1G, P_A1_BN1B, P_A1_W1, P_A1_BN2G, P_A1_BN2B, P_A1_W2,
  P_A2_BN1G, P_A2_BN1B, P_A2_W1, P_A2_BN2G, P_A2_BN2B, P_A2_W2,
  P_R0_W1, P_R0_B1, P_R0_W2, P_R0_B2, P_R0_CW, P_R0_CB,
  P_R1_W1, P_R1_B1, P_R1_W2, P_R1_B2, P_R1_CW, P_R1_CB,
  P_R2_W1, P_R2_B1, P_R2_W2, P_R2_B2, P_R2_CW, P_R2_CB,
  P_G_W1, P_G_B1, P_G_W2, P_G_B2,
  P_O_W, P_O_B, P_M_W1, P_M_B1, P_M_W2, P_M_B2
};

// Workspace layout (floats), with reuse of the big regions.
constexpr size_t OFF_SAMPD   = 0;                            // 51200
constexpr size_t OFF_COARSED = OFF_SAMPD   + 51200;          // 25600
constexpr size_t OFF_SAMPSEG = OFF_COARSED + 25600;          // 76800
constexpr size_t OFF_GEOME   = OFF_SAMPSEG + 76800;          // 15000
constexpr size_t OFF_CDTMP   = OFF_GEOME   + 15000;          // 25600
constexpr size_t OFF_CDP     = OFF_CDTMP   + 25600;          // 25600
constexpr size_t OFF_G2      = OFF_CDP     + 25600;          // 1280000
constexpr size_t OFF_H1F     = OFF_G2      + 1280000;        // 160000
constexpr size_t OFF_H2F     = OFF_H1F     + 160000;         // 40000
constexpr size_t OFF_RELTMP  = OFF_H2F     + 40000;          // 3840000
constexpr size_t OFF_RELS    = OFF_RELTMP  + 3840000;        // 2560000
constexpr size_t OFF_RELA    = OFF_RELS    + 2560000;        // 7680000
constexpr size_t OFF_BIGA    = OFF_RELA    + 7680000;        // 15360000 (Y1, then h1)
constexpr size_t OFF_BIGB    = OFF_BIGA    + 15360000;       //  9830400 (segmap, then rela_geom)
constexpr size_t OFF_BIGC    = OFF_BIGB    + 9830400;        //  7680000 (samp_rel, then g1)

extern "C" void kernel_launch(void* const* d_in, const int* in_sizes, int n_in,
                              void* d_out, int out_size, void* d_ws, size_t ws_size,
                              hipStream_t stream) {
  (void)in_sizes; (void)n_in; (void)out_size; (void)ws_size;
  auto F = [&](int i) { return (const float*)d_in[i]; };
  float* ws = (float*)d_ws;

  float* sampD   = ws + OFF_SAMPD;
  float* coarseD = ws + OFF_COARSED;
  float* sampSeg = ws + OFF_SAMPSEG;
  float* geome   = ws + OFF_GEOME;
  float* cdTmp   = ws + OFF_CDTMP;
  float* cdp     = ws + OFF_CDP;
  float* G2      = ws + OFF_G2;
  float* h1f     = ws + OFF_H1F;
  float* h2f     = ws + OFF_H2F;
  float* relTmp  = ws + OFF_RELTMP;
  float* relS    = ws + OFF_RELS;
  float* relaB   = ws + OFF_RELA;
  float* bigA    = ws + OFF_BIGA;   // Y1 (38400x256) then h1 (B,512,15000)
  float* bigB    = ws + OFF_BIGB;   // segmap (B,C,HW) then rela_geom (B,C,15000)
  float* bigC    = ws + OFF_BIGC;   // samp_rel (B,C,15000) then g1 (B,C,15000)

  const float s_bn1 = 1.0f / sqrtf(1.0f + 1.1e-5f);
  const float s_bn2 = 1.0f / sqrtf(1.0f + 1.0e-5f);
  auto blk = [](int n) { return dim3((unsigned)((n + 255) / 256)); };

  // 1) nearest sampling (depth + global_pos, coarse += old_points_depth)
  k_sample_nearest<<<blk(Bq * Cq * 150), 256, 0, stream>>>(
      F(IN_DEPTH), F(IN_GPOS), F(IN_REFER), F(IN_COARSE), F(IN_OLDD),
      sampD, coarseD);

  // 2) seg MLP: X(38400,256) -> gelu -> Y1 -> seg_map (B,C,H,W)
  gemm<1, 0, 0, 1>(stream, F(IN_SEG), F(P_SEG_W1), F(P_SEG_B1),
                   nullptr, nullptr, 0.0f, bigA, Bq * HWq, Cq, Cq, HWq, 1);
  gemm<0, 0, 1, 0>(stream, bigA, F(P_SEG_W2), F(P_SEG_B2),
                   nullptr, nullptr, 0.0f, bigB, Bq * HWq, Cq, Cq, 1, HWq);

  // 3) bilinear sampling of seg_map at grid_cat; outer product samp_rel
  k_sample_bilinear<<<blk(Bq * Cq * 150), 256, 0, stream>>>(
      bigB, F(IN_REFER), F(IN_COARSE), sampSeg);
  k_samp_rel<<<blk(Bq * Cq * SPq), 256, 0, stream>>>(sampSeg, sampD, bigC);

  // 4) ASPP branches: rela = atrous(x, a1, 6) + atrous(x, a2, 12)
  gemm<1, 1, 1, 0>(stream, bigC, F(P_A1_W1), nullptr,
                   F(P_A1_BN1G), F(P_A1_BN1B), s_bn1, bigA,
                   Bq * SPq, 2 * Cq, Cq, SPq, SPq);
  conv3x3(stream, bigA, F(P_A1_W2), F(P_A1_BN2G), F(P_A1_BN2B), s_bn2,
          nullptr, relaB, 2 * Cq, Cq, 150, 100, 6, Bq, /*acc=*/0, /*act=*/0);
  gemm<1, 1, 1, 0>(stream, bigC, F(P_A2_W1), nullptr,
                   F(P_A2_BN1G), F(P_A2_BN1B), s_bn1, bigA,
                   Bq * SPq, 2 * Cq, Cq, SPq, SPq);
  conv3x3(stream, bigA, F(P_A2_W2), F(P_A2_BN2G), F(P_A2_BN2B), s_bn2,
          nullptr, relaB, 2 * Cq, Cq, 150, 100, 12, Bq, /*acc=*/1, /*act=*/0);

  // 5) rel stages: MLP over last axis (transposed view) + 3x3 conv + gelu
  const int rw1[3] = {P_R0_W1, P_R1_W1, P_R2_W1};
  const int rb1[3] = {P_R0_B1, P_R1_B1, P_R2_B1};
  const int rw2[3] = {P_R0_W2, P_R1_W2, P_R2_W2};
  const int rb2[3] = {P_R0_B2, P_R1_B2, P_R2_B2};
  const int rcw[3] = {P_R0_CW, P_R1_CW, P_R2_CW};
  const int rcb[3] = {P_R0_CB, P_R1_CB, P_R2_CB};
  for (int si = 0; si < 3; ++si) {
    const int Kdim = (si == 0) ? 150 : 50;
    const int hid  = (si == 0) ? 75 : 50;
    gemm<1, 0, 0, 1>(stream, relaB, F(rw1[si]), F(rb1[si]),
                     nullptr, nullptr, 0.0f, relTmp,
                     Bq * Cq * 100, hid, Kdim, 100, 1);
    gemm<0, 0, 1, 0>(stream, relTmp, F(rw2[si]), F(rb2[si]),
                     nullptr, nullptr, 0.0f, relS,
                     Bq * Cq * 100, Lq, hid, 1, 100);
    conv3x3(stream, relS, F(rcw[si]), nullptr, nullptr, 0.0f,
            F(rcb[si]), relaB, Cq, Cq, 50, 100, 1, Bq, /*acc=*/0, /*act=*/1);
  }

  // 6) geometry (dist + sides) and coarse-depth MLP
  k_geometry<<<blk(Bq * Lq * 3 * Lq), 256, 0, stream>>>(
      F(IN_REFER), F(IN_COARSE), geome);
  gemm<0, 0, 0, 1>(stream, coarseD, F(P_CD_W1), F(P_CD_B1),
                   nullptr, nullptr, 0.0f, cdTmp, Bq * Cq, Lq, Lq, 1, 1);
  gemm<0, 0, 0, 0>(stream, cdTmp, F(P_CD_W2), F(P_CD_B2),
                   nullptr, nullptr, 0.0f, cdp, Bq * Cq, Lq, Lq, 1, 1);

  // 7) rela_geom assembly (B,C,300,50)
  k_rela_geom<<<blk(Bq * Cq * 300 * 50), 256, 0, stream>>>(
      relaB, geome, cdp, bigB);

  // 8) g = gelu(1x1 conv) then (6,1)-stride conv, packed straight into G2
  gemm<1, 1, 1, 1>(stream, bigB, F(P_G_W1), F(P_G_B1),
                   nullptr, nullptr, 0.0f, bigC, Bq * SPq, Cq, Cq, SPq, SPq);
  gemm<2, 1, 2, 0>(stream, bigC, F(P_G_W2), F(P_G_B2),
                   nullptr, nullptr, 0.0f, G2, Bq * 2500, Cq, Cq * 6, 1, 1);

  // 9) output head: gelu(G2 @ o_w + o_b) -> mlp -> sigmoid
  gemm<0, 0, 0, 1>(stream, G2, F(P_O_W), F(P_O_B),
                   nullptr, nullptr, 0.0f, h1f, Bq * Lq, 1600, 12800, 1, 1);
  gemm<0, 0, 0, 1>(stream, h1f, F(P_M_W1), F(P_M_B1),
                   nullptr, nullptr, 0.0f, h2f, Bq * Lq, 400, 1600, 1, 1);
  gemm<0, 0, 0, 2>(stream, h2f, F(P_M_W2), F(P_M_B2),
                   nullptr, nullptr, 0.0f, (float*)d_out, Bq * Lq, 1, 400, 1, 1);
}
